// RewardPredictionModel_12489764896949
// MI455X (gfx1250) — matrix operand order repaired and checked by reading, not compiled, not run on hardware
//
#include <hip/hip_runtime.h>
#include <hip/hip_bf16.h>

#define LATENT 512
#define BATCH  2048
#define SEQ    64
#define PRED   20
#define NEG_SLOPE 0.01f

typedef __bf16 v16bf __attribute__((ext_vector_type(16)));
typedef float  v8f   __attribute__((ext_vector_type(8)));
typedef float  v4f   __attribute__((ext_vector_type(4)));
typedef short  v8s   __attribute__((ext_vector_type(8)));

union FragBF { v16bf v; v8s h[2]; };

// ---- A fragment (16x32, bf16 source), ISA 7.12.2 "16-bit A-Matrix 16x32":
// lanes 0-15: VGPR0-3 <- K 0..7, VGPR4-7 <- K 16..23 ; lanes 16-31: +8
__device__ __forceinline__ v16bf load_a_bf16(const __bf16* base, int row0, int lda, int k0) {
  const int lane = threadIdx.x & 31;
  const int m    = lane & 15;
  const int half = lane >> 4;
  const __bf16* p = base + (size_t)(row0 + m) * lda + (k0 + half * 8);
  FragBF u;
  u.h[0] = *(const v8s*)(p);        // K = k0+koff .. +7
  u.h[1] = *(const v8s*)(p + 16);   // K = k0+koff+16 .. +23
  return u.v;
}

// ---- A fragment from fp32 source with on-the-fly bf16 convert (v_cvt_pk_bf16_f32)
__device__ __forceinline__ v16bf load_a_f32(const float* base, int row0, int lda, int k0) {
  const int lane = threadIdx.x & 31;
  const int m    = lane & 15;
  const int half = lane >> 4;
  const float* p = base + (size_t)(row0 + m) * lda + (k0 + half * 8);
  v4f f0 = *(const v4f*)(p);
  v4f f1 = *(const v4f*)(p + 4);
  v4f f2 = *(const v4f*)(p + 16);
  v4f f3 = *(const v4f*)(p + 20);
  v16bf v;
#pragma unroll
  for (int i = 0; i < 4; ++i) {
    v[i]      = (__bf16)f0[i];
    v[4 + i]  = (__bf16)f1[i];
    v[8 + i]  = (__bf16)f2[i];
    v[12 + i] = (__bf16)f3[i];
  }
  return v;
}

// ---- B fragment (32x16) from row-major W[N][K]: B[k][n] = W[n][k]
// lanes 0-15 hold K=0..7,16..23; lanes 16-31 hold K=8..15,24..31
__device__ __forceinline__ v16bf load_b(const __bf16* W, int n0, int ldw, int k0) {
  const int lane = threadIdx.x & 31;
  const int n    = lane & 15;
  const int half = lane >> 4;
  const __bf16* p = W + (size_t)(n0 + n) * ldw + (k0 + half * 8);
  FragBF u;
  u.h[0] = *(const v8s*)(p);       // K = k0+koff .. +7
  u.h[1] = *(const v8s*)(p + 16);  // K = k0+koff+16 .. +23
  return u.v;
}

__device__ __forceinline__ v8f wmma_bf16(v16bf a, v16bf b, v8f c) {
  return __builtin_amdgcn_wmma_f32_16x16x32_bf16(false, a, false, b, (short)0, c, false, false);
}

// One K=32 chunk: 2 A fragments x 4 gate B fragments -> 8 WMMAs
__device__ __forceinline__ void gemm_chunk(v16bf a0, v16bf a1,
                                           const __bf16* __restrict__ W, int ldw,
                                           int kb, int ncol, v8f acc[4][2]) {
#pragma unroll
  for (int g = 0; g < 4; ++g) {
    v16bf b = load_b(W, g * LATENT + ncol, ldw, kb);
    acc[g][0] = wmma_bf16(a0, b, acc[g][0]);
    acc[g][1] = wmma_bf16(a1, b, acc[g][1]);
  }
}

__device__ __forceinline__ float sigmoidf_(float x) { return 1.0f / (1.0f + __expf(-x)); }

// Fused LSTM cell step: gates GEMM (bf16 WMMA, fp32 acc) + elementwise epilogue.
// Workgroup: 8 waves -> 128-row x 32-col tile per gate; wave = 32 rows x 16 cols per gate.
template <bool ENC>
__global__ __launch_bounds__(256) void lstm_step_kernel(
    const float* __restrict__ x, int t,
    const __bf16* __restrict__ hbf_in,
    const __bf16* __restrict__ W, int K,
    const float* __restrict__ bsum,
    float* __restrict__ cf,
    __bf16* __restrict__ hbf_out) {
  const int wave = threadIdx.x >> 5;
  const int lane = threadIdx.x & 31;
  const int r    = wave & 3;          // row subtile (32 rows each)
  const int cc   = wave >> 2;         // col subtile (16 cols each)
  const int brow = blockIdx.y * 128 + r * 32;
  const int ncol = blockIdx.x * 32 + cc * 16;

  v8f acc[4][2] = {};   // [gate i,f,g,o][row tile 0,1]

  if (ENC) {
    // Phase 1: x_t contribution (fp32 source, on-the-fly bf16), K' = 0..511
    const float* xt = x + (size_t)t * LATENT;
    for (int k0 = 0; k0 < LATENT; k0 += 32) {
      v16bf a0 = load_a_f32(xt, brow,      SEQ * LATENT, k0);
      v16bf a1 = load_a_f32(xt, brow + 16, SEQ * LATENT, k0);
      gemm_chunk(a0, a1, W, K, k0, ncol, acc);
    }
    // Phase 2: h contribution (bf16 source), K' = 512..1023
    for (int k0 = 0; k0 < LATENT; k0 += 32) {
      v16bf a0 = load_a_bf16(hbf_in, brow,      LATENT, k0);
      v16bf a1 = load_a_bf16(hbf_in, brow + 16, LATENT, k0);
      gemm_chunk(a0, a1, W, K, LATENT + k0, ncol, acc);
    }
  } else {
    // Decode: x = h  ->  gates = h @ (Wih+Whh)^T, K = 512
    for (int k0 = 0; k0 < K; k0 += 32) {
      v16bf a0 = load_a_bf16(hbf_in, brow,      LATENT, k0);
      v16bf a1 = load_a_bf16(hbf_in, brow + 16, LATENT, k0);
      gemm_chunk(a0, a1, W, K, k0, ncol, acc);
    }
  }

  // Epilogue. C/D layout (32-bit 16x16): VGPR j -> M = j + 8*half, N = lane&15
  const int   half = lane >> 4;
  const int   col  = ncol + (lane & 15);
  const float bi = bsum[col];
  const float bf = bsum[col + LATENT];
  const float bg = bsum[col + 2 * LATENT];
  const float bo = bsum[col + 3 * LATENT];
#pragma unroll
  for (int rt = 0; rt < 2; ++rt) {
#pragma unroll
    for (int j = 0; j < 8; ++j) {
      const int    row = brow + rt * 16 + j + half * 8;
      const size_t idx = (size_t)row * LATENT + col;
      const float iv = sigmoidf_(acc[0][rt][j] + bi);
      const float fv = sigmoidf_(acc[1][rt][j] + bf);
      const float gv = tanhf(acc[2][rt][j] + bg);
      const float ov = sigmoidf_(acc[3][rt][j] + bo);
      const float cn = fv * cf[idx] + iv * gv;
      cf[idx]      = cn;
      hbf_out[idx] = (__bf16)(ov * tanhf(cn));
    }
  }
}

// Head: one wave32 per batch row; lane j = neuron j; shfl for the 32x32 layer.
__global__ __launch_bounds__(256) void head_kernel(
    const __bf16* __restrict__ hbf,
    const float* __restrict__ W1, const float* __restrict__ b1,
    const float* __restrict__ W2, const float* __restrict__ b2,
    const float* __restrict__ W3, const float* __restrict__ b3,
    float* __restrict__ out, int p) {
  const int lane = threadIdx.x & 31;
  const int row  = (int)((blockIdx.x * blockDim.x + threadIdx.x) >> 5);
  if (row >= BATCH) return;
  const __bf16* h  = hbf + (size_t)row * LATENT;
  const float*  w1 = W1 + (size_t)lane * LATENT;

  float a1 = 0.f;
  for (int k = 0; k < LATENT; k += 8) {
    FragBF u; u.h[0] = *(const v8s*)(h + k);
#pragma unroll
    for (int i = 0; i < 8; ++i) a1 += (float)u.v[i] * w1[k + i];
  }
  a1 += b1[lane];
  a1 = a1 > 0.f ? a1 : NEG_SLOPE * a1;

  float a2 = 0.f;
#pragma unroll
  for (int i = 0; i < 32; ++i) a2 += __shfl(a1, i, 32) * W2[lane * 32 + i];
  a2 += b2[lane];
  a2 = a2 > 0.f ? a2 : NEG_SLOPE * a2;

  float partial = a2 * W3[lane];
#pragma unroll
  for (int off = 16; off > 0; off >>= 1) partial += __shfl_down(partial, off, 32);
  if (lane == 0) out[(size_t)row * PRED + p] = partial + b3[0];
}

// Prep: Wcat = bf16([Wih | Whh]) [4H][1024]; Wdec = bf16(Wih+Whh) [4H][512]; bsum = bih+bhh
__global__ void prep_weights_kernel(const float* __restrict__ Wih, const float* __restrict__ Whh,
                                    const float* __restrict__ bih, const float* __restrict__ bhh,
                                    __bf16* __restrict__ Wcat, __bf16* __restrict__ Wdec,
                                    float* __restrict__ bsum) {
  const int idx = blockIdx.x * blockDim.x + threadIdx.x;
  if (idx >= 4 * LATENT * 2 * LATENT) return;
  const int n = idx >> 10;     // / 1024
  const int k = idx & 1023;
  const float v = (k < LATENT) ? Wih[(size_t)n * LATENT + k]
                               : Whh[(size_t)n * LATENT + (k - LATENT)];
  Wcat[idx] = (__bf16)v;
  if (k < LATENT)
    Wdec[(size_t)n * LATENT + k] =
        (__bf16)(Wih[(size_t)n * LATENT + k] + Whh[(size_t)n * LATENT + k]);
  if (idx < 4 * LATENT) bsum[idx] = bih[idx] + bhh[idx];
}

__global__ void init_state_kernel(float* __restrict__ cf, __bf16* __restrict__ h0) {
  const int idx = blockIdx.x * blockDim.x + threadIdx.x;
  if (idx >= BATCH * LATENT) return;
  cf[idx] = 0.f;
  h0[idx] = (__bf16)0.f;
}

extern "C" void kernel_launch(void* const* d_in, const int* in_sizes, int n_in,
                              void* d_out, int out_size, void* d_ws, size_t ws_size,
                              hipStream_t stream) {
  (void)in_sizes; (void)n_in; (void)out_size; (void)ws_size;
  const float* x   = (const float*)d_in[0];
  const float* Wih = (const float*)d_in[1];
  const float* Whh = (const float*)d_in[2];
  const float* bih = (const float*)d_in[3];
  const float* bhh = (const float*)d_in[4];
  const float* W1  = (const float*)d_in[5];
  const float* b1  = (const float*)d_in[6];
  const float* W2  = (const float*)d_in[7];
  const float* b2  = (const float*)d_in[8];
  const float* W3  = (const float*)d_in[9];
  const float* b3  = (const float*)d_in[10];
  float* out = (float*)d_out;

  char*  ws  = (char*)d_ws;
  size_t off = 0;
  auto alloc = [&](size_t bytes) {
    char* p = ws + off;
    off += (bytes + 255) & ~(size_t)255;
    return p;
  };
  __bf16* Wcat = (__bf16*)alloc((size_t)4 * LATENT * 2 * LATENT * sizeof(__bf16)); // 4 MB
  __bf16* Wdec = (__bf16*)alloc((size_t)4 * LATENT * LATENT * sizeof(__bf16));     // 2 MB
  float*  bsum = (float*)alloc((size_t)4 * LATENT * sizeof(float));
  float*  cf   = (float*)alloc((size_t)BATCH * LATENT * sizeof(float));            // 4 MB
  __bf16* hA   = (__bf16*)alloc((size_t)BATCH * LATENT * sizeof(__bf16));          // 2 MB
  __bf16* hB   = (__bf16*)alloc((size_t)BATCH * LATENT * sizeof(__bf16));          // 2 MB

  {
    int n = 4 * LATENT * 2 * LATENT;
    prep_weights_kernel<<<(n + 255) / 256, 256, 0, stream>>>(Wih, Whh, bih, bhh, Wcat, Wdec, bsum);
  }
  {
    int n = BATCH * LATENT;
    init_state_kernel<<<(n + 255) / 256, 256, 0, stream>>>(cf, hA);
  }

  dim3 grid(LATENT / 32, BATCH / 128);   // 16 x 16 workgroups of 8 waves
  __bf16* cur = hA;
  __bf16* nxt = hB;
  for (int t = 0; t < SEQ; ++t) {
    lstm_step_kernel<true><<<grid, 256, 0, stream>>>(x, t, cur, Wcat, 2 * LATENT, bsum, cf, nxt);
    __bf16* tmp = cur; cur = nxt; nxt = tmp;
  }
  for (int p = 0; p < PRED; ++p) {
    lstm_step_kernel<false><<<grid, 256, 0, stream>>>(nullptr, 0, cur, Wdec, LATENT, bsum, cf, nxt);
    __bf16* tmp = cur; cur = nxt; nxt = tmp;
    head_kernel<<<BATCH / 8, 256, 0, stream>>>(cur, W1, b1, W2, b2, W3, b3, out, p);
  }
}
